// MultiHeadAttention_54168127537535
// MI455X (gfx1250) — matrix operand order repaired and checked by reading, not compiled, not run on hardware
//
#include <hip/hip_runtime.h>
#include <stdint.h>

// ---------------------------------------------------------------------------
// MI455X (gfx1250) MHA forward, bf16 WMMA pipeline + async LDS staging.
// B=2, S=2048, D=1024, H=16, d_k=64.
// ---------------------------------------------------------------------------

typedef __attribute__((ext_vector_type(16))) __bf16 v16bf;
typedef __attribute__((ext_vector_type(8)))  float  v8f;

// Pointer types matching __builtin_amdgcn_global_load_async_to_lds_b128's
// signature (int __vector(4) __device__* / __shared__*).
typedef int v4i_vs __attribute__((vector_size(16)));
typedef __attribute__((address_space(1))) v4i_vs* gbl_v4i_ptr;
typedef __attribute__((address_space(3))) v4i_vs* lds_v4i_ptr;

#define BATCH   2
#define SEQ     2048
#define DMODEL  1024
#define NHEADS  16
#define DK      64
#define MTOT    (BATCH * SEQ)          // 4096 rows

#if defined(__gfx1250__) && __has_builtin(__builtin_amdgcn_global_load_async_to_lds_b128)
#define HAVE_ASYNC_LDS 1
#else
#define HAVE_ASYNC_LDS 0
#endif

__device__ __forceinline__ void wait_asynccnt0() {
#if defined(__gfx1250__)
#if __has_builtin(__builtin_amdgcn_s_wait_asynccnt)
    __builtin_amdgcn_s_wait_asynccnt(0);
#else
    asm volatile("s_wait_asynccnt 0x0" ::: "memory");
#endif
#endif
}

// A-fragment (16x32 bf16) K index for element e, half-wave h:
//   K = e + 16h
__device__ __forceinline__ int a_kidx(int e, int half) {
    return e + 16 * half;
}

// ---------------------------------------------------------------------------
// Kernel 0: fp32 -> bf16 convert
// ---------------------------------------------------------------------------
__global__ void __launch_bounds__(256)
cvt_f32_bf16(const float* __restrict__ src, __bf16* __restrict__ dst, int n) {
    int i = blockIdx.x * 256 + threadIdx.x;
    if (i < n) dst[i] = (__bf16)src[i];
}

// ---------------------------------------------------------------------------
// Kernel 1: projection GEMM  Y = X @ W^T   (X:[4096,1024] bf16, W:[1024,1024] bf16)
// Output stored head-major bf16: Yh[((b*16 + h)*2048 + s)*64 + dk]
// One wave computes a 32(M) x 64(N) tile: 2 A-frags x 4 B-frags = 8 WMMAs/k-step.
// ---------------------------------------------------------------------------
__global__ void __launch_bounds__(128)
proj_qkv_kernel(const __bf16* __restrict__ X, const __bf16* __restrict__ W,
                __bf16* __restrict__ Yh) {
    const int wid  = threadIdx.x >> 5;
    const int lane = threadIdx.x & 31;
    const int strip = blockIdx.x * 4 + wid;          // 2048 strips total
    const int mtile = strip & 127;                    // M/32 = 128
    const int ntile = strip >> 7;                     // N/64 = 16 (== head)
    const int mrow  = mtile * 32;
    const int nbase = ntile * 64;
    const int r    = lane & 15;
    const int half = lane >> 4;

    v8f acc[2][4] = {};
    for (int kb = 0; kb < DMODEL; kb += 32) {
        if (kb + 64 < DMODEL) {
            __builtin_prefetch(&X[(size_t)(mrow + r) * DMODEL + kb + 64], 0, 0);
            __builtin_prefetch(&W[(size_t)(nbase + r) * DMODEL + kb + 64], 0, 0);
        }
        v16bf a0, a1;
        #pragma unroll
        for (int e = 0; e < 16; ++e) {
            a0[e] = X[(size_t)(mrow + r) * DMODEL + kb + a_kidx(e, half)];
            a1[e] = X[(size_t)(mrow + 16 + r) * DMODEL + kb + a_kidx(e, half)];
        }
        #pragma unroll
        for (int nt = 0; nt < 4; ++nt) {
            v16bf bfrag;
            #pragma unroll
            for (int e = 0; e < 16; ++e)
                bfrag[e] = W[(size_t)(nbase + nt * 16 + r) * DMODEL + kb + e + 16 * half];
            acc[0][nt] = __builtin_amdgcn_wmma_f32_16x16x32_bf16(
                false, a0, false, bfrag, (short)0, acc[0][nt], false, false);
            acc[1][nt] = __builtin_amdgcn_wmma_f32_16x16x32_bf16(
                false, a1, false, bfrag, (short)0, acc[1][nt], false, false);
        }
    }
    // store: C/D layout M = i + 8*half, N-lane = r
    #pragma unroll
    for (int mt = 0; mt < 2; ++mt) {
        #pragma unroll
        for (int nt = 0; nt < 4; ++nt) {
            #pragma unroll
            for (int i = 0; i < 8; ++i) {
                int m  = mrow + mt * 16 + i + 8 * half;
                int bb = m >> 11;               // m / SEQ
                int s  = m & (SEQ - 1);
                int dk = nt * 16 + r;
                Yh[(((size_t)(bb * NHEADS + ntile)) * SEQ + s) * DK + dk] =
                    (__bf16)acc[mt][nt][i];
            }
        }
    }
}

// ---------------------------------------------------------------------------
// Kernel 2: attention. One block per (b, h, 16-row q-tile).
// 256 threads = 8 waves.
// Dynamic LDS (160KB -> 2 blocks/WGP):
//   [0,128K)    : scores 16x2048 f32
//   [128K,160K) : mask bytes 16x2048 (async-staged); aliased by ctx 16x64 f32
//                 after softmax (mask is dead by then).
// ---------------------------------------------------------------------------
__global__ void __launch_bounds__(256)
attention_kernel(const __bf16* __restrict__ Qh, const __bf16* __restrict__ Kh,
                 const __bf16* __restrict__ Vh, const uint8_t* __restrict__ mask,
                 float* __restrict__ attn, __bf16* __restrict__ Ctxb) {
    extern __shared__ char smemraw[];
    float*         sc    = (float*)smemraw;                    // [16][2048] f32
    unsigned char* smask = (unsigned char*)(smemraw + 16 * SEQ * 4); // [16][2048] u8
    float*         ctxs  = (float*)(smemraw + 16 * SEQ * 4);   // [16][64] f32 (aliases smask)

    const int qtile = blockIdx.x;        // S/16 = 128
    const int h     = blockIdx.y;
    const int b     = blockIdx.z;
    const int qbase = qtile * 16;
    const int wid   = threadIdx.x >> 5;
    const int lane  = threadIdx.x & 31;
    const int r     = lane & 15;
    const int half  = lane >> 4;
    const size_t headoff = (size_t)(b * NHEADS + h) * SEQ * DK;

    // ---- stage 32KB contiguous mask chunk for this q-tile into LDS (async) ----
    const uint8_t* gmask = mask + ((size_t)b * SEQ + qbase) * SEQ;
#if HAVE_ASYNC_LDS
    #pragma unroll
    for (int it = 0; it < 8; ++it) {
        int off = (it * 256 + (int)threadIdx.x) * 16;
        __builtin_amdgcn_global_load_async_to_lds_b128(
            (gbl_v4i_ptr)(uintptr_t)(gmask + off),
            (lds_v4i_ptr)(uint32_t)(uintptr_t)(smask + off),
            0, 0);
    }
#else
    for (int off = threadIdx.x * 4; off < 16 * SEQ; off += 256 * 4)
        *(uint32_t*)(smask + off) = *(const uint32_t*)(gmask + off);
#endif

    // ---- Phase 1: raw scores = (Q @ K^T) / 8 -> LDS (overlaps mask copy) ----
    for (int j = 0; j < 16; ++j) {
        int colbase = (wid * 16 + j) * 16;           // key tile base
        v8f acc = {};
        #pragma unroll
        for (int kb = 0; kb < DK; kb += 32) {
            v16bf a, bfrag;
            #pragma unroll
            for (int e = 0; e < 16; ++e) {
                a[e]     = Qh[headoff + (size_t)(qbase + r) * DK + kb + a_kidx(e, half)];
                bfrag[e] = Kh[headoff + (size_t)(colbase + r) * DK + kb + e + 16 * half];
            }
            acc = __builtin_amdgcn_wmma_f32_16x16x32_bf16(
                false, a, false, bfrag, (short)0, acc, false, false);
        }
        #pragma unroll
        for (int i = 0; i < 8; ++i) {
            int m   = i + 8 * half;                   // row within q-tile
            int col = colbase + r;
            sc[m * SEQ + col] = acc[i] * 0.125f;      // 1/sqrt(64)
        }
    }
    wait_asynccnt0();
    __syncthreads();

    // ---- Phase 2: apply mask + softmax per row; write normalized attn ----
    #pragma unroll
    for (int rr = 0; rr < 2; ++rr) {
        int row = wid * 2 + rr;
        const unsigned char* mrow_ = smask + row * SEQ;
        float mx = -1e30f;
        for (int c = lane; c < SEQ; c += 32) {
            float v = sc[row * SEQ + c];
            if (mrow_[c]) v = -1e9f;
            sc[row * SEQ + c] = v;
            mx = fmaxf(mx, v);
        }
        #pragma unroll
        for (int o = 16; o > 0; o >>= 1) mx = fmaxf(mx, __shfl_xor(mx, o));
        float sum = 0.0f;
        for (int c = lane; c < SEQ; c += 32) sum += __expf(sc[row * SEQ + c] - mx);
        #pragma unroll
        for (int o = 16; o > 0; o >>= 1) sum += __shfl_xor(sum, o);
        float inv = 1.0f / sum;
        size_t arow = ((size_t)(b * NHEADS + h) * SEQ + qbase + row) * SEQ;
        for (int c = lane; c < SEQ; c += 32) {
            float p = __expf(sc[row * SEQ + c] - mx) * inv;
            sc[row * SEQ + c] = p;
            attn[arow + c] = p;
        }
    }
    __syncthreads();

    // mask region is dead; reuse first 4KB as context accumulator
    for (int idx = threadIdx.x; idx < 16 * DK; idx += 256) ctxs[idx] = 0.0f;
    __syncthreads();

    // ---- Phase 3: context = P @ V. Each wave covers 256 keys x full 64 cols ----
    {
        v8f acc[4] = {};
        for (int kk = 0; kk < 8; ++kk) {
            int kb = wid * 256 + kk * 32;
            v16bf a;
            #pragma unroll
            for (int e = 0; e < 16; ++e)
                a[e] = (__bf16)sc[r * SEQ + kb + a_kidx(e, half)];
            #pragma unroll
            for (int nt = 0; nt < 4; ++nt) {
                v16bf bfrag;
                #pragma unroll
                for (int e = 0; e < 16; ++e)
                    bfrag[e] = Vh[headoff + (size_t)(kb + e + 16 * half) * DK + nt * 16 + r];
                acc[nt] = __builtin_amdgcn_wmma_f32_16x16x32_bf16(
                    false, a, false, bfrag, (short)0, acc[nt], false, false);
            }
        }
        #pragma unroll
        for (int nt = 0; nt < 4; ++nt)
            #pragma unroll
            for (int i = 0; i < 8; ++i)
                atomicAdd(&ctxs[(i + 8 * half) * DK + nt * 16 + r], acc[nt][i]);
    }
    __syncthreads();

    // ---- write context tile, interleaved back to [B,S,D] bf16 ----
    for (int idx = threadIdx.x; idx < 16 * DK; idx += 256) {
        int row = idx >> 6, dk = idx & 63;
        Ctxb[((size_t)b * SEQ + qbase + row) * DMODEL + h * DK + dk] = (__bf16)ctxs[idx];
    }
}

// ---------------------------------------------------------------------------
// Kernel 3: output projection + residual:  Xres = Ctx @ Wfc^T + input_Q  (fp32)
// 32x64 tile per wave, same scheme as proj_qkv.
// ---------------------------------------------------------------------------
__global__ void __launch_bounds__(128)
proj_out_kernel(const __bf16* __restrict__ X, const __bf16* __restrict__ W,
                const float* __restrict__ resid, float* __restrict__ Y) {
    const int wid  = threadIdx.x >> 5;
    const int lane = threadIdx.x & 31;
    const int strip = blockIdx.x * 4 + wid;
    const int mtile = strip & 127;
    const int ntile = strip >> 7;
    const int mrow  = mtile * 32;
    const int nbase = ntile * 64;
    const int r    = lane & 15;
    const int half = lane >> 4;

    v8f acc[2][4] = {};
    for (int kb = 0; kb < DMODEL; kb += 32) {
        if (kb + 64 < DMODEL) {
            __builtin_prefetch(&X[(size_t)(mrow + r) * DMODEL + kb + 64], 0, 0);
            __builtin_prefetch(&W[(size_t)(nbase + r) * DMODEL + kb + 64], 0, 0);
        }
        v16bf a0, a1;
        #pragma unroll
        for (int e = 0; e < 16; ++e) {
            a0[e] = X[(size_t)(mrow + r) * DMODEL + kb + a_kidx(e, half)];
            a1[e] = X[(size_t)(mrow + 16 + r) * DMODEL + kb + a_kidx(e, half)];
        }
        #pragma unroll
        for (int nt = 0; nt < 4; ++nt) {
            v16bf bfrag;
            #pragma unroll
            for (int e = 0; e < 16; ++e)
                bfrag[e] = W[(size_t)(nbase + nt * 16 + r) * DMODEL + kb + e + 16 * half];
            acc[0][nt] = __builtin_amdgcn_wmma_f32_16x16x32_bf16(
                false, a0, false, bfrag, (short)0, acc[0][nt], false, false);
            acc[1][nt] = __builtin_amdgcn_wmma_f32_16x16x32_bf16(
                false, a1, false, bfrag, (short)0, acc[1][nt], false, false);
        }
    }
    #pragma unroll
    for (int mt = 0; mt < 2; ++mt) {
        #pragma unroll
        for (int nt = 0; nt < 4; ++nt) {
            #pragma unroll
            for (int i = 0; i < 8; ++i) {
                int m = mrow + mt * 16 + i + 8 * half;
                int n = nbase + nt * 16 + r;
                size_t idx = (size_t)m * DMODEL + n;
                Y[idx] = acc[mt][nt][i] + resid[idx];
            }
        }
    }
}

// ---------------------------------------------------------------------------
// Kernel 4: LayerNorm (gamma=1, beta=0, biased variance, eps=1e-5)
// ---------------------------------------------------------------------------
__global__ void __launch_bounds__(256)
ln_kernel(const float* __restrict__ X, float* __restrict__ O) {
    __shared__ float rs[8], rs2[8];
    const int row = blockIdx.x;
    const float* x = X + (size_t)row * DMODEL;
    float s = 0.0f, s2 = 0.0f;
    for (int i = threadIdx.x; i < DMODEL; i += 256) {
        float v = x[i]; s += v; s2 += v * v;
    }
    #pragma unroll
    for (int o = 16; o > 0; o >>= 1) { s += __shfl_xor(s, o); s2 += __shfl_xor(s2, o); }
    int wid = threadIdx.x >> 5, lane = threadIdx.x & 31;
    if (lane == 0) { rs[wid] = s; rs2[wid] = s2; }
    __syncthreads();
    if (wid == 0) {
        s  = (lane < 8) ? rs[lane]  : 0.0f;
        s2 = (lane < 8) ? rs2[lane] : 0.0f;
        #pragma unroll
        for (int o = 4; o > 0; o >>= 1) { s += __shfl_xor(s, o); s2 += __shfl_xor(s2, o); }
        if (lane == 0) { rs[0] = s; rs2[0] = s2; }
    }
    __syncthreads();
    float mean = rs[0] * (1.0f / DMODEL);
    float var  = rs2[0] * (1.0f / DMODEL) - mean * mean;
    float rinv = rsqrtf(var + 1e-5f);
    for (int i = threadIdx.x; i < DMODEL; i += 256)
        O[(size_t)row * DMODEL + i] = (x[i] - mean) * rinv;
}

// ---------------------------------------------------------------------------
// Host launcher
// ---------------------------------------------------------------------------
extern "C" void kernel_launch(void* const* d_in, const int* in_sizes, int n_in,
                              void* d_out, int out_size, void* d_ws, size_t ws_size,
                              hipStream_t stream) {
    const float*   inQ  = (const float*)d_in[0];
    const float*   inK  = (const float*)d_in[1];
    const float*   inV  = (const float*)d_in[2];
    const uint8_t* mask = (const uint8_t*)d_in[3];
    const float*   Wq   = (const float*)d_in[4];
    const float*   Wk   = (const float*)d_in[5];
    const float*   Wv   = (const float*)d_in[6];
    const float*   Wfc  = (const float*)d_in[7];

    char* ws = (char*)d_ws;
    const size_t MB = 1024 * 1024;
    __bf16* Xq   = (__bf16*)(ws + 0 * MB);    // [4096,1024] bf16, 8MB
    __bf16* Xk   = (__bf16*)(ws + 8 * MB);
    __bf16* Xv   = (__bf16*)(ws + 16 * MB);
    __bf16* Wqb  = (__bf16*)(ws + 24 * MB);   // [1024,1024] bf16, 2MB each
    __bf16* Wkb  = (__bf16*)(ws + 26 * MB);
    __bf16* Wvb  = (__bf16*)(ws + 28 * MB);
    __bf16* Wfcb = (__bf16*)(ws + 30 * MB);
    __bf16* Qh   = (__bf16*)(ws + 32 * MB);   // [B,H,S,64] bf16, 8MB each
    __bf16* Kh   = (__bf16*)(ws + 40 * MB);
    __bf16* Vh   = (__bf16*)(ws + 48 * MB);
    __bf16* Ctxb = (__bf16*)(ws + 56 * MB);   // [B,S,D] bf16, 8MB
    float*  Xres = (float*) (ws + 64 * MB);   // [B,S,D] f32, 16MB

    float* out  = (float*)d_out;
    float* attn = out + (size_t)MTOT * DMODEL;  // 4,194,304 offset

    const int nIn = MTOT * DMODEL;     // 4,194,304
    const int nW  = DMODEL * DMODEL;   // 1,048,576

    cvt_f32_bf16<<<(nIn + 255) / 256, 256, 0, stream>>>(inQ, Xq, nIn);
    cvt_f32_bf16<<<(nIn + 255) / 256, 256, 0, stream>>>(inK, Xk, nIn);
    cvt_f32_bf16<<<(nIn + 255) / 256, 256, 0, stream>>>(inV, Xv, nIn);
    cvt_f32_bf16<<<(nW  + 255) / 256, 256, 0, stream>>>(Wq,  Wqb, nW);
    cvt_f32_bf16<<<(nW  + 255) / 256, 256, 0, stream>>>(Wk,  Wkb, nW);
    cvt_f32_bf16<<<(nW  + 255) / 256, 256, 0, stream>>>(Wv,  Wvb, nW);
    cvt_f32_bf16<<<(nW  + 255) / 256, 256, 0, stream>>>(Wfc, Wfcb, nW);

    // 2048 strips / 4 waves per block
    proj_qkv_kernel<<<512, 128, 0, stream>>>(Xq, Wqb, Qh);
    proj_qkv_kernel<<<512, 128, 0, stream>>>(Xk, Wkb, Kh);
    proj_qkv_kernel<<<512, 128, 0, stream>>>(Xv, Wvb, Vh);

    // LDS: scores 128KB + mask 32KB (ctx aliased) = 160KB -> 2 blocks per WGP
    size_t lds = (size_t)16 * SEQ * 4 + (size_t)16 * SEQ;
    attention_kernel<<<dim3(SEQ / 16, NHEADS, BATCH), 256, lds, stream>>>(
        Qh, Kh, Vh, mask, attn, Ctxb);

    proj_out_kernel<<<512, 128, 0, stream>>>(Ctxb, Wfcb, inQ, Xres);
    ln_kernel<<<MTOT, 256, 0, stream>>>(Xres, out);
}
